// LLaDAExpert_11665131176296
// MI455X (gfx1250) — matrix operand order repaired and checked by reading, not compiled, not run on hardware
//
#include <hip/hip_runtime.h>
#include <hip/hip_bf16.h>
#include <stdint.h>

// Problem constants (match reference)
#define K_D 2048          // N_EMBD
#define K_H 4096          // HIDDEN
#define K_A 256           // ADAPT
#define K_S 2048          // S
#define K_B 4             // B
#define K_M (K_B * K_S)   // 8192 rows total

typedef __attribute__((ext_vector_type(16))) __bf16 bf16x16;
typedef __attribute__((ext_vector_type(8)))  float  floatx8;

union FragBF {
  uint4   q[2];   // 32 bytes = 16 bf16
  bf16x16 v;
};

__device__ __forceinline__ uint16_t f2bf(float f) {
  uint32_t u = __float_as_uint(f);
  u += 0x7FFFu + ((u >> 16) & 1u);   // round-to-nearest-even
  return (uint16_t)(u >> 16);
}

// silu with native reciprocal (v_rcp_f32) instead of IEEE div chain
__device__ __forceinline__ float silu_f(float x) {
  return x * __builtin_amdgcn_rcpf(1.0f + __expf(-x));
}

// ---------------------------------------------------------------------------
// f32 -> bf16 cast, 4 elements/thread (float4 in, 2x packed u32 out)
// ---------------------------------------------------------------------------
__global__ void cast_f32_to_bf16(const float4* __restrict__ in,
                                 uint2* __restrict__ out, int n4) {
  int i = blockIdx.x * blockDim.x + threadIdx.x;
  if (i < n4) {
    float4 f = in[i];
    uint2 p;
    p.x = (uint32_t)f2bf(f.x) | ((uint32_t)f2bf(f.y) << 16);
    p.y = (uint32_t)f2bf(f.z) | ((uint32_t)f2bf(f.w) << 16);
    out[i] = p;
  }
}

// ---------------------------------------------------------------------------
// Wave-level WMMA mainloop, 2x2 tile: rows {m, m+16} x cols {n, n+16}.
// 4 WMMAs per 8 b128 loads per k-step.
// A: [M,K] row-major bf16.  Bw: [N,K] row-major bf16 (i.e. B transposed).
// Fragment layouts used here (wave32, contiguous k = 16*khalf + s for A and B):
//   A 16x32 bf16: lane = (m%16) + 16*khalf; two contiguous 16B loads
//   B 32x16 bf16: lane = (n%16) + 16*kgroup; two contiguous 16B loads
// ---------------------------------------------------------------------------
__device__ __forceinline__ void gemm_accum4(const uint16_t* __restrict__ Amat, int lda,
                                            const uint16_t* __restrict__ Bw,  int ldb,
                                            int K, int rowA, int colB, int lane,
                                            floatx8 (&acc)[2][2]) {
  const int r = lane & 15;
  const int h = lane >> 4;
  const uint16_t* ap0 = Amat + (size_t)(rowA + r) * lda + h * 16;
  const uint16_t* ap1 = ap0 + (size_t)16 * lda;
  const uint16_t* bp0 = Bw  + (size_t)(colB + r) * ldb + h * 16;
  const uint16_t* bp1 = bp0 + (size_t)16 * ldb;
  __builtin_prefetch(ap0, 0, 3);
  __builtin_prefetch(ap1, 0, 3);
  __builtin_prefetch(bp0, 0, 3);
  __builtin_prefetch(bp1, 0, 3);
#pragma unroll 2
  for (int kb = 0; kb < K; kb += 32) {
    FragBF a0, a1, b0, b1;
    a0.q[0] = *(const uint4*)(ap0);
    a0.q[1] = *(const uint4*)(ap0 + 8);
    a1.q[0] = *(const uint4*)(ap1);
    a1.q[1] = *(const uint4*)(ap1 + 8);
    b0.q[0] = *(const uint4*)(bp0);
    b0.q[1] = *(const uint4*)(bp0 + 8);
    b1.q[0] = *(const uint4*)(bp1);
    b1.q[1] = *(const uint4*)(bp1 + 8);
    acc[0][0] = __builtin_amdgcn_wmma_f32_16x16x32_bf16(false, a0.v, false, b0.v,
                                                        (short)0, acc[0][0], false, false);
    acc[1][0] = __builtin_amdgcn_wmma_f32_16x16x32_bf16(false, a1.v, false, b0.v,
                                                        (short)0, acc[1][0], false, false);
    acc[0][1] = __builtin_amdgcn_wmma_f32_16x16x32_bf16(false, a0.v, false, b1.v,
                                                        (short)0, acc[0][1], false, false);
    acc[1][1] = __builtin_amdgcn_wmma_f32_16x16x32_bf16(false, a1.v, false, b1.v,
                                                        (short)0, acc[1][1], false, false);
    ap0 += 32; ap1 += 32; bp0 += 32; bp1 += 32;
  }
}

// ---------------------------------------------------------------------------
// Trunk: hidden = silu(x @ Wg^T) * (x @ Wu^T)
// grid (M/32, H/256), 256 threads (8 waves).
// Per wave: 2 M-tiles x 2 N-tiles x 2 weight chains -> 8 WMMAs per 12 loads.
// ---------------------------------------------------------------------------
__global__ void trunk_kernel(const uint16_t* __restrict__ xb,
                             const uint16_t* __restrict__ wub,
                             const uint16_t* __restrict__ wgb,
                             float* __restrict__ hidden_f32,
                             uint16_t* __restrict__ hidden_bf) {
  const int wave = threadIdx.x >> 5;
  const int lane = threadIdx.x & 31;
  const int m = blockIdx.x * 32;
  const int n = blockIdx.y * 256 + wave * 32;
  const int r = lane & 15;
  const int h = lane >> 4;

  floatx8 up[2][2] = {};
  floatx8 gt[2][2] = {};

  const uint16_t* ap0 = xb  + (size_t)(m + r) * K_D + h * 16;
  const uint16_t* ap1 = ap0 + (size_t)16 * K_D;
  const uint16_t* bu0 = wub + (size_t)(n + r) * K_D + h * 16;
  const uint16_t* bu1 = bu0 + (size_t)16 * K_D;
  const uint16_t* bg0 = wgb + (size_t)(n + r) * K_D + h * 16;
  const uint16_t* bg1 = bg0 + (size_t)16 * K_D;
  __builtin_prefetch(ap0, 0, 3);
  __builtin_prefetch(bu0, 0, 3);
  __builtin_prefetch(bg0, 0, 3);
#pragma unroll 1
  for (int kb = 0; kb < K_D; kb += 32) {
    FragBF a0, a1, u0, u1, g0, g1;
    a0.q[0] = *(const uint4*)(ap0);
    a0.q[1] = *(const uint4*)(ap0 + 8);
    a1.q[0] = *(const uint4*)(ap1);
    a1.q[1] = *(const uint4*)(ap1 + 8);
    u0.q[0] = *(const uint4*)(bu0);
    u0.q[1] = *(const uint4*)(bu0 + 8);
    u1.q[0] = *(const uint4*)(bu0 + 16 * K_D);
    u1.q[1] = *(const uint4*)(bu0 + 16 * K_D + 8);
    g0.q[0] = *(const uint4*)(bg0);
    g0.q[1] = *(const uint4*)(bg0 + 8);
    g1.q[0] = *(const uint4*)(bg0 + 16 * K_D);
    g1.q[1] = *(const uint4*)(bg0 + 16 * K_D + 8);
    up[0][0] = __builtin_amdgcn_wmma_f32_16x16x32_bf16(false, a0.v, false, u0.v,
                                                       (short)0, up[0][0], false, false);
    up[1][0] = __builtin_amdgcn_wmma_f32_16x16x32_bf16(false, a1.v, false, u0.v,
                                                       (short)0, up[1][0], false, false);
    up[0][1] = __builtin_amdgcn_wmma_f32_16x16x32_bf16(false, a0.v, false, u1.v,
                                                       (short)0, up[0][1], false, false);
    up[1][1] = __builtin_amdgcn_wmma_f32_16x16x32_bf16(false, a1.v, false, u1.v,
                                                       (short)0, up[1][1], false, false);
    gt[0][0] = __builtin_amdgcn_wmma_f32_16x16x32_bf16(false, a0.v, false, g0.v,
                                                       (short)0, gt[0][0], false, false);
    gt[1][0] = __builtin_amdgcn_wmma_f32_16x16x32_bf16(false, a1.v, false, g0.v,
                                                       (short)0, gt[1][0], false, false);
    gt[0][1] = __builtin_amdgcn_wmma_f32_16x16x32_bf16(false, a0.v, false, g1.v,
                                                       (short)0, gt[0][1], false, false);
    gt[1][1] = __builtin_amdgcn_wmma_f32_16x16x32_bf16(false, a1.v, false, g1.v,
                                                       (short)0, gt[1][1], false, false);
    ap0 += 32; ap1 += 32; bu0 += 32; bg0 += 32;
  }
  (void)bu1; (void)bg1;

#pragma unroll
  for (int j = 0; j < 2; ++j) {
#pragma unroll
    for (int c = 0; c < 2; ++c) {
      const int col  = n + c * 16 + r;
      const int row0 = m + j * 16 + h * 8;
#pragma unroll
      for (int i = 0; i < 8; ++i) {
        float hv = silu_f(gt[j][c][i]) * up[j][c][i];
        size_t idx = (size_t)(row0 + i) * K_H + col;
        hidden_f32[idx] = hv;
        hidden_bf[idx]  = f2bf(hv);
      }
    }
  }
}

// ---------------------------------------------------------------------------
// Generic bf16 GEMM -> f32 output (adapter projections before LayerNorm)
// grid (M/32, N/256)
// ---------------------------------------------------------------------------
__global__ void gemm_f32_out(const uint16_t* __restrict__ Amat, int lda,
                             const uint16_t* __restrict__ Bw, int ldb,
                             float* __restrict__ C, int ldc, int K) {
  const int wave = threadIdx.x >> 5;
  const int lane = threadIdx.x & 31;
  const int m = blockIdx.x * 32;
  const int n = blockIdx.y * 256 + wave * 32;
  floatx8 acc[2][2] = {};
  gemm_accum4(Amat, lda, Bw, ldb, K, m, n, lane, acc);
  const int r  = lane & 15;
  const int h8 = (lane >> 4) * 8;
#pragma unroll
  for (int c = 0; c < 2; ++c) {
    const int col = n + c * 16 + r;
#pragma unroll
    for (int i = 0; i < 8; ++i) {
      C[(size_t)(m + h8 + i) * ldc + col]      = acc[0][c][i];
      C[(size_t)(m + 16 + h8 + i) * ldc + col] = acc[1][c][i];
    }
  }
}

// ---------------------------------------------------------------------------
// LayerNorm over last dim (256), one row per block, 256 threads.
// ---------------------------------------------------------------------------
__global__ void ln_kernel(const float* __restrict__ in,
                          const float* __restrict__ g,
                          const float* __restrict__ b,
                          uint16_t* __restrict__ out) {
  __shared__ float red[256];
  const int row = blockIdx.x;
  const int t   = threadIdx.x;
  float v = in[(size_t)row * K_A + t];

  red[t] = v;
  __syncthreads();
#pragma unroll
  for (int s = 128; s > 0; s >>= 1) {
    if (t < s) red[t] += red[t + s];
    __syncthreads();
  }
  const float mu = red[0] * (1.0f / K_A);
  __syncthreads();

  const float d = v - mu;
  red[t] = d * d;
  __syncthreads();
#pragma unroll
  for (int s = 128; s > 0; s >>= 1) {
    if (t < s) red[t] += red[t + s];
    __syncthreads();
  }
  const float var = red[0] * (1.0f / K_A);
  const float y = d * rsqrtf(var + 1e-5f) * g[t] + b[t];
  out[(size_t)row * K_A + t] = f2bf(y);
}

// ---------------------------------------------------------------------------
// Per-batch transpose [S, A] -> [A, S] (bf16) via 32x32 LDS tiles.
// grid (S/32, A/32, B), 256 threads.
// ---------------------------------------------------------------------------
__global__ void transpose_kernel(const uint16_t* __restrict__ in,
                                 uint16_t* __restrict__ out) {
  __shared__ uint16_t tile[32][33];
  const int tx = threadIdx.x & 31;
  const int ty = threadIdx.x >> 5;   // 0..7
  const uint16_t* src = in  + (size_t)blockIdx.z * K_S * K_A;
  uint16_t*       dst = out + (size_t)blockIdx.z * K_S * K_A;

#pragma unroll
  for (int i = 0; i < 32; i += 8) {
    int s = blockIdx.x * 32 + ty + i;
    int a = blockIdx.y * 32 + tx;
    tile[ty + i][tx] = src[(size_t)s * K_A + a];
  }
  __syncthreads();
#pragma unroll
  for (int i = 0; i < 32; i += 8) {
    int a = blockIdx.y * 32 + ty + i;
    int s = blockIdx.x * 32 + tx;
    dst[(size_t)a * K_S + s] = tile[tx][ty + i];
  }
}

// ---------------------------------------------------------------------------
// aw = silu(clip(adapt_in @ adapt_out^T, +-5))  per batch, bf16 out.
// grid (S/32, S/256, B)
// ---------------------------------------------------------------------------
__global__ void aw_kernel(const uint16_t* __restrict__ ain,
                          const uint16_t* __restrict__ aout,
                          uint16_t* __restrict__ aw) {
  const int wave = threadIdx.x >> 5;
  const int lane = threadIdx.x & 31;
  const int bb = blockIdx.z;
  const int m = blockIdx.x * 32;
  const int n = blockIdx.y * 256 + wave * 32;
  floatx8 acc[2][2] = {};
  gemm_accum4(ain + (size_t)bb * K_S * K_A, K_A,
              aout + (size_t)bb * K_S * K_A, K_A, K_A, m, n, lane, acc);
  const int r  = lane & 15;
  const int h8 = (lane >> 4) * 8;
  uint16_t* dst = aw + (size_t)bb * K_S * K_S;
#pragma unroll
  for (int c = 0; c < 2; ++c) {
    const int col = n + c * 16 + r;
#pragma unroll
    for (int i = 0; i < 8; ++i) {
      float v0 = fminf(5.0f, fmaxf(-5.0f, acc[0][c][i]));
      float v1 = fminf(5.0f, fmaxf(-5.0f, acc[1][c][i]));
      dst[(size_t)(m + h8 + i) * K_S + col]      = f2bf(silu_f(v0));
      dst[(size_t)(m + 16 + h8 + i) * K_S + col] = f2bf(silu_f(v1));
    }
  }
}

// ---------------------------------------------------------------------------
// adapt = aw @ adapt_in  per batch (uses transposed adapt_in), bf16 out.
// grid (S/32, A/256 (=1), B)
// ---------------------------------------------------------------------------
__global__ void adapt_kernel(const uint16_t* __restrict__ aw,
                             const uint16_t* __restrict__ ainT,
                             uint16_t* __restrict__ adapt) {
  const int wave = threadIdx.x >> 5;
  const int lane = threadIdx.x & 31;
  const int bb = blockIdx.z;
  const int m = blockIdx.x * 32;
  const int n = wave * 32;
  floatx8 acc[2][2] = {};
  gemm_accum4(aw + (size_t)bb * K_S * K_S, K_S,
              ainT + (size_t)bb * K_A * K_S, K_S, K_S, m, n, lane, acc);
  const int r  = lane & 15;
  const int h8 = (lane >> 4) * 8;
  uint16_t* dst = adapt + (size_t)bb * K_S * K_A;
#pragma unroll
  for (int c = 0; c < 2; ++c) {
    const int col = n + c * 16 + r;
#pragma unroll
    for (int i = 0; i < 8; ++i) {
      dst[(size_t)(m + h8 + i) * K_A + col]      = f2bf(acc[0][c][i]);
      dst[(size_t)(m + 16 + h8 + i) * K_A + col] = f2bf(acc[1][c][i]);
    }
  }
}

// ---------------------------------------------------------------------------
// hidden2 = bf16( hidden_f32 + 0.1 * (adapt @ Wproj^T) )
// grid (M/32, H/256)
// ---------------------------------------------------------------------------
__global__ void proj_kernel(const uint16_t* __restrict__ adapt,
                            const uint16_t* __restrict__ wprojb,
                            const float* __restrict__ hidden_f32,
                            uint16_t* __restrict__ hidden2_bf) {
  const int wave = threadIdx.x >> 5;
  const int lane = threadIdx.x & 31;
  const int m = blockIdx.x * 32;
  const int n = blockIdx.y * 256 + wave * 32;
  floatx8 acc[2][2] = {};
  gemm_accum4(adapt, K_A, wprojb, K_A, K_A, m, n, lane, acc);
  const int r  = lane & 15;
  const int h8 = (lane >> 4) * 8;
#pragma unroll
  for (int c = 0; c < 2; ++c) {
    const int col = n + c * 16 + r;
#pragma unroll
    for (int i = 0; i < 8; ++i) {
      size_t i0 = (size_t)(m + h8 + i) * K_H + col;
      size_t i1 = (size_t)(m + 16 + h8 + i) * K_H + col;
      hidden2_bf[i0] = f2bf(hidden_f32[i0] + 0.1f * acc[0][c][i]);
      hidden2_bf[i1] = f2bf(hidden_f32[i1] + 0.1f * acc[1][c][i]);
    }
  }
}

// ---------------------------------------------------------------------------
// out = hidden2 @ Wdown^T   (f32 output)
// grid (M/32, D/256)
// ---------------------------------------------------------------------------
__global__ void down_kernel(const uint16_t* __restrict__ hidden2_bf,
                            const uint16_t* __restrict__ wdb,
                            float* __restrict__ out) {
  const int wave = threadIdx.x >> 5;
  const int lane = threadIdx.x & 31;
  const int m = blockIdx.x * 32;
  const int n = blockIdx.y * 256 + wave * 32;
  floatx8 acc[2][2] = {};
  gemm_accum4(hidden2_bf, K_H, wdb, K_H, K_H, m, n, lane, acc);
  const int r  = lane & 15;
  const int h8 = (lane >> 4) * 8;
#pragma unroll
  for (int c = 0; c < 2; ++c) {
    const int col = n + c * 16 + r;
#pragma unroll
    for (int i = 0; i < 8; ++i) {
      out[(size_t)(m + h8 + i) * K_D + col]      = acc[0][c][i];
      out[(size_t)(m + 16 + h8 + i) * K_D + col] = acc[1][c][i];
    }
  }
}

// ---------------------------------------------------------------------------
// Host launcher
// ---------------------------------------------------------------------------
extern "C" void kernel_launch(void* const* d_in, const int* in_sizes, int n_in,
                              void* d_out, int out_size, void* d_ws, size_t ws_size,
                              hipStream_t stream) {
  const float* x      = (const float*)d_in[0];
  const float* W_up   = (const float*)d_in[1];
  const float* W_gate = (const float*)d_in[2];
  const float* W_down = (const float*)d_in[3];
  const float* W_pre  = (const float*)d_in[4];
  const float* W_post = (const float*)d_in[5];
  const float* W_proj = (const float*)d_in[6];
  const float* ln_g   = (const float*)d_in[7];
  const float* ln_b   = (const float*)d_in[8];
  float* out = (float*)d_out;

  char* wptr = (char*)d_ws;
  auto alloc = [&](size_t bytes) -> void* {
    void* p = (void*)wptr;
    wptr += (bytes + 255) & ~(size_t)255;
    return p;
  };

  uint16_t* xb      = (uint16_t*)alloc((size_t)K_M * K_D * 2);
  uint16_t* wub     = (uint16_t*)alloc((size_t)K_H * K_D * 2);
  uint16_t* wgb     = (uint16_t*)alloc((size_t)K_H * K_D * 2);
  uint16_t* wdb     = (uint16_t*)alloc((size_t)K_D * K_H * 2);
  uint16_t* wpreb   = (uint16_t*)alloc((size_t)K_A * K_D * 2);
  uint16_t* wpostb  = (uint16_t*)alloc((size_t)K_A * K_H * 2);
  uint16_t* wprojb  = (uint16_t*)alloc((size_t)K_H * K_A * 2);
  float*    hid_f32 = (float*)   alloc((size_t)K_M * K_H * 4);
  uint16_t* hid_bf  = (uint16_t*)alloc((size_t)K_M * K_H * 2);
  float*    pre_f32 = (float*)   alloc((size_t)K_M * K_A * 4);
  float*    post_f32= (float*)   alloc((size_t)K_M * K_A * 4);
  uint16_t* ain_bf  = (uint16_t*)alloc((size_t)K_M * K_A * 2);
  uint16_t* aout_bf = (uint16_t*)alloc((size_t)K_M * K_A * 2);
  uint16_t* ainT_bf = (uint16_t*)alloc((size_t)K_M * K_A * 2);
  uint16_t* aw_bf   = (uint16_t*)alloc((size_t)K_B * K_S * K_S * 2);
  uint16_t* adapt_bf= (uint16_t*)alloc((size_t)K_M * K_A * 2);
  uint16_t* hid2_bf = (uint16_t*)alloc((size_t)K_M * K_H * 2);

  const int TB = 256;
  auto castN = [&](const float* src, uint16_t* dst, size_t n) {
    size_t n4 = n / 4;
    cast_f32_to_bf16<<<dim3((unsigned)((n4 + TB - 1) / TB)), dim3(TB), 0, stream>>>(
        (const float4*)src, (uint2*)dst, (int)n4);
  };

  // 1) casts
  castN(x,      xb,     (size_t)K_M * K_D);
  castN(W_up,   wub,    (size_t)K_H * K_D);
  castN(W_gate, wgb,    (size_t)K_H * K_D);
  castN(W_down, wdb,    (size_t)K_D * K_H);
  castN(W_pre,  wpreb,  (size_t)K_A * K_D);
  castN(W_post, wpostb, (size_t)K_A * K_H);
  castN(W_proj, wprojb, (size_t)K_H * K_A);

  // 2) trunk: hidden = silu(gate) * up
  trunk_kernel<<<dim3(K_M / 32, K_H / 256), dim3(TB), 0, stream>>>(
      xb, wub, wgb, hid_f32, hid_bf);

  // 3) adapt_in = LN(x @ Wpre^T)
  gemm_f32_out<<<dim3(K_M / 32, K_A / 256), dim3(TB), 0, stream>>>(
      xb, K_D, wpreb, K_D, pre_f32, K_A, K_D);
  ln_kernel<<<dim3(K_M), dim3(TB), 0, stream>>>(pre_f32, ln_g, ln_b, ain_bf);

  // 4) adapt_out = LN(hidden @ Wpost^T)
  gemm_f32_out<<<dim3(K_M / 32, K_A / 256), dim3(TB), 0, stream>>>(
      hid_bf, K_H, wpostb, K_H, post_f32, K_A, K_H);
  ln_kernel<<<dim3(K_M), dim3(TB), 0, stream>>>(post_f32, ln_g, ln_b, aout_bf);

  // 5) transpose adapt_in per batch for the second adapter GEMM
  transpose_kernel<<<dim3(K_S / 32, K_A / 32, K_B), dim3(TB), 0, stream>>>(
      ain_bf, ainT_bf);

  // 6) aw = silu(clip(adapt_in @ adapt_out^T))
  aw_kernel<<<dim3(K_S / 32, K_S / 256, K_B), dim3(TB), 0, stream>>>(
      ain_bf, aout_bf, aw_bf);

  // 7) adapt = aw @ adapt_in
  adapt_kernel<<<dim3(K_S / 32, 1, K_B), dim3(TB), 0, stream>>>(
      aw_bf, ainT_bf, adapt_bf);

  // 8) hidden2 = hidden + 0.1 * adapt @ Wproj^T
  proj_kernel<<<dim3(K_M / 32, K_H / 256), dim3(TB), 0, stream>>>(
      adapt_bf, wprojb, hid_f32, hid2_bf);

  // 9) out = hidden2 @ Wdown^T
  down_kernel<<<dim3(K_M / 32, K_D / 256), dim3(TB), 0, stream>>>(
      hid2_bf, wdb, out);

  (void)in_sizes; (void)n_in; (void)out_size; (void)ws_size;
}
